// MorphoXPro_3324304687587
// MI455X (gfx1250) — matrix operand order repaired and checked by reading, not compiled, not run on hardware
//
#include <hip/hip_runtime.h>
#include <hip/hip_bf16.h>

// ---------------------------------------------------------------------------
// MorphoXPro fused implementation for MI455X (gfx1250, wave32, WMMA).
//
// Dominant cost: per-layer router matmul h[B,32] @ r_w2[32, dout*din]
// fused with sigmoid-mask * weight * x reduction (the 268MB logits tensor is
// never materialized). V_WMMA_F32_16X16X32_BF16, K=32 == router hidden dim.
//
// Assumed d_in order (setup_inputs dict insertion order):
//   0: x[256,512]
//   per layer l in {0,1} (13 tensors each, starting at 1 and 14):
//     +0 r_w1[din,32] +1 r_b1[32] +2 r_w2[32,dout*din] +3 r_b2[dout*din]
//     +4 mask_bias[dout,din] +5 weight[dout,din] +6 bias[dout]
//     +7 gp_global[5] +8 g_w1[dout,32] +9 g_b1[32] +10 g_w2[32,5]
//     +11 g_b2[5] +12 blend[1]
//   27: o_w1[256,128] 28: o_b1[128] 29: o_w2[128,10] 30: o_b2[10]
// Output: [256,10] float32.
// ---------------------------------------------------------------------------

typedef __attribute__((ext_vector_type(16))) __bf16 v16bf;
typedef __attribute__((ext_vector_type(8)))  __bf16 v8bf;
typedef __attribute__((ext_vector_type(8)))  float  v8f;

#define BATCH   256
#define RH      32      // router hidden
#define NPRIM   5
#define WAVES_PER_BLOCK 4

// ---------------------------------------------------------------------------
// Generic small dense: out[m,n] = act(X[m,:K] . W[:K,n] + b[n]); optionally
// also emits bf16 copy (for the WMMA A-matrix operand).
// ---------------------------------------------------------------------------
__global__ void dense_kernel(const float* __restrict__ X,
                             const float* __restrict__ W,
                             const float* __restrict__ bias,
                             float* __restrict__ out,
                             __bf16* __restrict__ out_bf,
                             int M, int K, int N, int do_relu) {
    int idx = blockIdx.x * blockDim.x + threadIdx.x;
    if (idx >= M * N) return;
    int m = idx / N, n = idx % N;
    float acc = bias ? bias[n] : 0.0f;
    const float* xr = X + (size_t)m * K;
    for (int k = 0; k < K; ++k)
        acc = fmaf(xr[k], W[(size_t)k * N + n], acc);
    if (do_relu) acc = fmaxf(acc, 0.0f);
    if (out)    out[idx]    = acc;
    if (out_bf) out_bf[idx] = (__bf16)acc;
}

// ---------------------------------------------------------------------------
// Transpose + convert r_w2 [32, cols] f32 (row-major) -> [cols, 32] bf16.
// Reads are coalesced across cols; writes are 4x b128 per thread.
// ---------------------------------------------------------------------------
__global__ void transpose_bf16_kernel(const float* __restrict__ w,
                                      __bf16* __restrict__ wt, int cols) {
    int col = blockIdx.x * blockDim.x + threadIdx.x;
    if (col >= cols) return;
    __bf16 tmp[32];
#pragma unroll
    for (int k = 0; k < 32; ++k)
        tmp[k] = (__bf16)w[(size_t)k * cols + col];
    v8bf* dst = (v8bf*)(wt + (size_t)col * 32);
    const v8bf* src = (const v8bf*)tmp;
    dst[0] = src[0]; dst[1] = src[1]; dst[2] = src[2]; dst[3] = src[3];
}

// ---------------------------------------------------------------------------
// Pre-fold the three per-column scalar streams into one float2 stream:
//   meta[c] = { r_b2[c] + mask_bias[c], weight[c] }   (identical flat index)
// Turns 3 dependent b32 loads + 1 add per tile-lane into one b64 load.
// ---------------------------------------------------------------------------
__global__ void combine_meta_kernel(const float* __restrict__ r_b2,
                                    const float* __restrict__ mask_bias,
                                    const float* __restrict__ weight,
                                    float2* __restrict__ meta, int cols) {
    int c = blockIdx.x * blockDim.x + threadIdx.x;
    if (c >= cols) return;
    meta[c] = make_float2(r_b2[c] + mask_bias[c], weight[c]);
}

// ---------------------------------------------------------------------------
// Fused masked-linear:
//   y[b,o] = bias[o] + sum_i x[b,i] * weight[o,i] *
//            sigmoid( (h[b,:] @ r_w2[:, o*din+i]) + r_b2[..] + mask_bias[..] )
// One wave per (16-row batch tile, output neuron o); DIN is a template
// constant (512 for both layers) so the 32-tile loop fully unrolls and every
// load is base-pointer + immediate IOFFSET (no per-iteration address math).
//
// VGPR layouts per CDNA5 ISA 7.12.2:
//   A (16x32 bf16): lanes 0-15 -> M=lane, K {0-7,16-23}; lanes 16-31 -> K {8-15,24-31}
//   B (32x16 bf16): lanes 0-15 -> N=lane, K {0-7,16-23}; lanes 16-31 -> K {8-15,24-31}
//   C (16x16 f32) : VGPR p, lanes 0-15 -> M=p, N=lane; lanes 16-31 -> M=p+8
// ---------------------------------------------------------------------------
template <int DIN>
__global__ void masked_linear_wmma_kernel(
    const float*  __restrict__ x,       // [BATCH, DIN]
    const __bf16* __restrict__ hbf,     // [BATCH, 32]   router hidden, bf16
    const __bf16* __restrict__ rw2t,    // [dout*DIN, 32] transposed bf16
    const float2* __restrict__ meta,    // [dout*DIN] {r_b2+mask_bias, weight}
    const float*  __restrict__ bias_,   // [dout]
    float*        __restrict__ y,       // [BATCH, dout]
    int dout) {
    constexpr int NT = DIN / 16;
    const int lane = threadIdx.x & 31;
    const int wave = threadIdx.x >> 5;
    const int n    = lane & 15;
    const int hi   = lane >> 4;
    const int b0   = blockIdx.x * 16;
    const int o    = blockIdx.y * WAVES_PER_BLOCK + wave;   // grid sized exactly

    // ---- A tile: h[b0..b0+15, 0..31] as bf16, ISA interleaved layout ----
    const __bf16* ar = hbf + (size_t)(b0 + n) * RH;
    v8bf a_lo = *(const v8bf*)(ar + hi * 8);        // K 0-7 (lo) / 8-15 (hi)
    v8bf a_hi = *(const v8bf*)(ar + 16 + hi * 8);   // K 16-23 (lo) / 24-31 (hi)
    v16bf A;
#pragma unroll
    for (int e = 0; e < 8; ++e) { A[e] = a_lo[e]; A[e + 8] = a_hi[e]; }

    // ---- hoisted base pointers: all loop loads are base + immediate ----
    const size_t colbase = (size_t)o * DIN;
    const __bf16* brow = rw2t + (colbase + n) * RH + hi * 8;   // += it*16*RH
    const float2* mrow = meta + colbase + n;                   // += it*16
    const float* xrow[8];
#pragma unroll
    for (int p = 0; p < 8; ++p)
        xrow[p] = x + (size_t)(b0 + p + hi * 8) * DIN + n;     // += it*16

    float acc[8] = {0.f, 0.f, 0.f, 0.f, 0.f, 0.f, 0.f, 0.f};

#pragma unroll
    for (int it = 0; it < NT; ++it) {
        v8bf b_lo = *(const v8bf*)(brow + it * 16 * RH);
        v8bf b_hi = *(const v8bf*)(brow + it * 16 * RH + 16);
        v16bf Bv;
#pragma unroll
        for (int e = 0; e < 8; ++e) { Bv[e] = b_lo[e]; Bv[e + 8] = b_hi[e]; }

        v8f c = {};
        c = __builtin_amdgcn_wmma_f32_16x16x32_bf16(
                /*neg_a=*/false, A, /*neg_b=*/false, Bv,
                /*c_mod=*/(short)0, c, /*reuse_a=*/false, /*reuse_b=*/false);

        const float2 mt = mrow[it * 16];
#pragma unroll
        for (int p = 0; p < 8; ++p) {
            const float logit = c[p] + mt.x;
            const float m = 1.0f / (1.0f + __expf(-logit));
            acc[p] = fmaf(xrow[p][it * 16] * mt.y, m, acc[p]);
        }
    }

    // reduce over the 16 columns (lanes within each half of the wave)
#pragma unroll
    for (int p = 0; p < 8; ++p) {
        float v = acc[p];
        v += __shfl_xor(v, 1, 32);
        v += __shfl_xor(v, 2, 32);
        v += __shfl_xor(v, 4, 32);
        v += __shfl_xor(v, 8, 32);
        acc[p] = v;
    }
    if (n == 0) {
        const float bo = bias_[o];
#pragma unroll
        for (int p = 0; p < 8; ++p)
            y[(size_t)(b0 + p + hi * 8) * dout + o] = acc[p] + bo;
    }
}

// ---------------------------------------------------------------------------
// Gated-primitive blend. Block per batch row; every thread recomputes the
// 5-way softmax weights (160 MACs, cheaper than another kernel round-trip).
// acts: [relu, exact gelu, tanh, silu, identity]
// ---------------------------------------------------------------------------
__global__ void gate_blend_kernel(const float* __restrict__ y,
                                  const float* __restrict__ gh,       // [B,32]
                                  const float* __restrict__ g_w2,     // [32,5]
                                  const float* __restrict__ g_b2,     // [5]
                                  const float* __restrict__ gp_global,// [5]
                                  const float* __restrict__ blend_p,  // [1]
                                  float* __restrict__ out, int d) {
    const int b = blockIdx.x;
    float gl[NPRIM];
#pragma unroll
    for (int j = 0; j < NPRIM; ++j) gl[j] = g_b2[j];
    const float* ghb = gh + (size_t)b * RH;
    for (int k = 0; k < RH; ++k) {
        const float g = ghb[k];
#pragma unroll
        for (int j = 0; j < NPRIM; ++j) gl[j] = fmaf(g, g_w2[k * NPRIM + j], gl[j]);
    }
    float mx = gl[0];
#pragma unroll
    for (int j = 1; j < NPRIM; ++j) mx = fmaxf(mx, gl[j]);
    float s = 0.f;
#pragma unroll
    for (int j = 0; j < NPRIM; ++j) { gl[j] = __expf(gl[j] - mx); s += gl[j]; }
    float gp[NPRIM];
    float mg = gp_global[0];
#pragma unroll
    for (int j = 1; j < NPRIM; ++j) mg = fmaxf(mg, gp_global[j]);
    float sg = 0.f;
#pragma unroll
    for (int j = 0; j < NPRIM; ++j) { gp[j] = __expf(gp_global[j] - mg); sg += gp[j]; }
    const float bl = *blend_p;
    float w[NPRIM];
#pragma unroll
    for (int j = 0; j < NPRIM; ++j)
        w[j] = (1.0f - bl) * (gp[j] / sg) + bl * (gl[j] / s);

    const float* yb = y + (size_t)b * d;
    float* ob = out + (size_t)b * d;
    for (int i = threadIdx.x; i < d; i += blockDim.x) {
        const float v  = yb[i];
        const float r  = fmaxf(v, 0.0f);
        const float ge = 0.5f * v * (1.0f + erff(v * 0.70710678118654752f));
        const float th = tanhf(v);
        const float si = v / (1.0f + __expf(-v));
        ob[i] = w[0] * r + w[1] * ge + w[2] * th + w[3] * si + w[4] * v;
    }
}

// ---------------------------------------------------------------------------
extern "C" void kernel_launch(void* const* d_in, const int* in_sizes, int n_in,
                              void* d_out, int out_size, void* d_ws, size_t ws_size,
                              hipStream_t stream) {
    (void)in_sizes; (void)n_in; (void)out_size; (void)ws_size;

    const float* x = (const float*)d_in[0];
    struct LayerP {
        const float *r_w1, *r_b1, *r_w2, *r_b2, *mask_bias, *weight, *bias;
        const float *gp, *g_w1, *g_b1, *g_w2, *g_b2, *blend;
    } L[2];
    int idx = 1;
    for (int l = 0; l < 2; ++l) {
        L[l].r_w1      = (const float*)d_in[idx++];
        L[l].r_b1      = (const float*)d_in[idx++];
        L[l].r_w2      = (const float*)d_in[idx++];
        L[l].r_b2      = (const float*)d_in[idx++];
        L[l].mask_bias = (const float*)d_in[idx++];
        L[l].weight    = (const float*)d_in[idx++];
        L[l].bias      = (const float*)d_in[idx++];
        L[l].gp        = (const float*)d_in[idx++];
        L[l].g_w1      = (const float*)d_in[idx++];
        L[l].g_b1      = (const float*)d_in[idx++];
        L[l].g_w2      = (const float*)d_in[idx++];
        L[l].g_b2      = (const float*)d_in[idx++];
        L[l].blend     = (const float*)d_in[idx++];
    }
    const float* o_w1 = (const float*)d_in[idx++];
    const float* o_b1 = (const float*)d_in[idx++];
    const float* o_w2 = (const float*)d_in[idx++];
    const float* o_b2 = (const float*)d_in[idx++];

    // ---- workspace layout (256B aligned) ----
    char* base = (char*)d_ws;
    size_t off = 0;
    auto carve = [&](size_t bytes) {
        char* p = base + off;
        off = (off + bytes + 255) & ~(size_t)255;
        return p;
    };
    __bf16* rw2t = (__bf16*)carve((size_t)512 * 512 * RH * sizeof(__bf16)); // 16.8MB, reused L1
    float2* meta = (float2*)carve((size_t)512 * 512 * sizeof(float2));      // 2MB, reused L1
    float*  h32  = (float*) carve((size_t)BATCH * RH * sizeof(float));
    __bf16* hbf  = (__bf16*)carve((size_t)BATCH * RH * sizeof(__bf16));
    float*  ybuf = (float*) carve((size_t)BATCH * 512 * sizeof(float));
    float*  gh   = (float*) carve((size_t)BATCH * RH * sizeof(float));
    float*  act0 = (float*) carve((size_t)BATCH * 512 * sizeof(float));
    float*  act1 = (float*) carve((size_t)BATCH * 256 * sizeof(float));
    float*  zbuf = (float*) carve((size_t)BATCH * 128 * sizeof(float));

    const int dims[3] = {512, 512, 256};
    const float* layer_in = x;
    float* layer_out[2] = {act0, act1};

    for (int l = 0; l < 2; ++l) {
        const int din = dims[l], dout = dims[l + 1];
        const int cols = dout * din;

        // router hidden: h = relu(in @ r_w1 + r_b1), plus bf16 copy for WMMA A
        dense_kernel<<<(BATCH * RH + 255) / 256, 256, 0, stream>>>(
            layer_in, L[l].r_w1, L[l].r_b1, h32, hbf, BATCH, din, RH, 1);

        // r_w2 -> [cols, 32] bf16 (WMMA B operand layout)
        transpose_bf16_kernel<<<(cols + 255) / 256, 256, 0, stream>>>(
            L[l].r_w2, rw2t, cols);

        // fold {r_b2 + mask_bias, weight} into one float2 stream
        combine_meta_kernel<<<(cols + 255) / 256, 256, 0, stream>>>(
            L[l].r_b2, L[l].mask_bias, L[l].weight, meta, cols);

        // fused masked linear (the WMMA kernel); din == 512 for both layers
        dim3 grid(BATCH / 16, dout / WAVES_PER_BLOCK);
        masked_linear_wmma_kernel<512><<<grid, 32 * WAVES_PER_BLOCK, 0, stream>>>(
            layer_in, hbf, rw2t, meta, L[l].bias, ybuf, dout);

        // gate hidden: gh = relu(y @ g_w1 + g_b1)
        dense_kernel<<<(BATCH * RH + 255) / 256, 256, 0, stream>>>(
            ybuf, L[l].g_w1, L[l].g_b1, gh, nullptr, BATCH, dout, RH, 1);

        // gated primitive blend -> next layer input
        gate_blend_kernel<<<BATCH, 256, 0, stream>>>(
            ybuf, gh, L[l].g_w2, L[l].g_b2, L[l].gp, L[l].blend,
            layer_out[l], dout);

        layer_in = layer_out[l];
    }

    // output head: z = relu(a2 @ o_w1 + o_b1); out = z @ o_w2 + o_b2
    dense_kernel<<<(BATCH * 128 + 255) / 256, 256, 0, stream>>>(
        act1, o_w1, o_b1, zbuf, nullptr, BATCH, 256, 128, 1);
    dense_kernel<<<(BATCH * 10 + 255) / 256, 256, 0, stream>>>(
        zbuf, o_w2, o_b2, (float*)d_out, nullptr, BATCH, 128, 10, 0);
}
